// SRUCell_74345883894034
// MI455X (gfx1250) — matrix operand order repaired and checked by reading, not compiled, not run on hardware
//
#include <hip/hip_runtime.h>
#include <math.h>

// ---------------------------------------------------------------------------
// SRU forward for MI455X (gfx1250, wave32, WMMA)
//   u = x @ W  (M=16384, K=2048, N=6144)  via bf16 WMMA, f32 accumulate
//   Pre-pass: x -> bf16 (same layout), W -> bf16 TRANSPOSED to Wt[N][K]
//   GEMM: BOTH A and B tiles streamed with global_load_async_to_lds_b128
//         (CDNA5 async DMA, ASYNCcnt), double-buffered LDS, 1 barrier/iter,
//         zero VGPR staging in the hot loop.
//   gates fused in GEMM epilogue (de-interleave N%3, sigmoid, bias)
//   scan kernel: c_t = (c_{t-1}-x~)*f + x~ ; h = (tanh(c)-x)*r + x
// ---------------------------------------------------------------------------

typedef __attribute__((ext_vector_type(8)))  __bf16         v8bf;
typedef __attribute__((ext_vector_type(16))) __bf16         v16bf;
typedef __attribute__((ext_vector_type(8)))  float          v8f;

constexpr int SEQ   = 512;
constexpr int BATCH = 32;
constexpr int DIM   = 2048;
constexpr int GM    = SEQ * BATCH;   // 16384 GEMM rows
constexpr int GN    = 3 * DIM;       // 6144  GEMM cols
constexpr int GK    = DIM;           // 2048  GEMM depth

constexpr int BM = 128;
constexpr int BN = 128;
constexpr int BK = 32;               // K step == WMMA K for bf16
constexpr int LDST = 40;             // LDS row stride (bf16): 80B, 16B aligned, bank-friendly
constexpr int KITERS = GK / BK;      // 64

__device__ __forceinline__ unsigned short f2bf(float x) {
  unsigned int u = __float_as_uint(x);
  u += 0x7FFFu + ((u >> 16) & 1u);   // round-to-nearest-even
  return (unsigned short)(u >> 16);
}

__device__ __forceinline__ float fast_sigmoid(float x) {
  return __builtin_amdgcn_rcpf(1.0f + __expf(-x));   // v_exp_f32 + v_rcp_f32
}
__device__ __forceinline__ float fast_tanh(float x) {
  return 1.0f - 2.0f * __builtin_amdgcn_rcpf(__expf(2.0f * x) + 1.0f);
}

// ---------------- x: f32 -> bf16, same layout (4 elems/thread) -------------
__global__ void f32_to_bf16_x4(const float* __restrict__ src,
                               unsigned short* __restrict__ dst, int n4) {
  int i = blockIdx.x * blockDim.x + threadIdx.x;
  if (i >= n4) return;
  const float4 v = ((const float4*)src)[i];
  unsigned int lo = (unsigned int)f2bf(v.x) | ((unsigned int)f2bf(v.y) << 16);
  unsigned int hi = (unsigned int)f2bf(v.z) | ((unsigned int)f2bf(v.w) << 16);
  ((uint2*)dst)[i] = make_uint2(lo, hi);
}

// ---------------- W: f32 [K][N] -> bf16 Wt [N][K] (LDS-tiled transpose) ----
constexpr int TT = 64;
__global__ void w_convert_transpose(const float* __restrict__ W,
                                    unsigned short* __restrict__ Wt) {
  __shared__ unsigned short T[TT][TT + 4];
  const int t = threadIdx.x;
  const int kBase = blockIdx.y * TT;
  const int nBase = blockIdx.x * TT;
#pragma unroll
  for (int p = 0; p < 4; ++p) {            // coalesced read along N
    const int k = p * 16 + (t >> 4);
    const int n = (t & 15) * 4;
    const float4 v = *(const float4*)(W + (size_t)(kBase + k) * GN + nBase + n);
    T[k][n + 0] = f2bf(v.x);
    T[k][n + 1] = f2bf(v.y);
    T[k][n + 2] = f2bf(v.z);
    T[k][n + 3] = f2bf(v.w);
  }
  __syncthreads();
#pragma unroll
  for (int p = 0; p < 4; ++p) {            // coalesced write along K
    const int n = p * 16 + (t >> 4);
    const int k = (t & 15) * 4;
    const unsigned a0 = T[k + 0][n], a1 = T[k + 1][n];
    const unsigned a2 = T[k + 2][n], a3 = T[k + 3][n];
    *(uint2*)(Wt + (size_t)(nBase + n) * GK + kBase + k) =
        make_uint2(a0 | (a1 << 16), a2 | (a3 << 16));
  }
}

// ---------------- bf16 WMMA GEMM + fused gate epilogue ---------------------
__launch_bounds__(256, 2)
__global__ void sru_gemm_bf16(const unsigned short* __restrict__ A,   // x_bf16 [GM][GK]
                              const unsigned short* __restrict__ Bt,  // Wt bf16 [GN][GK]
                              const float* __restrict__ bias,         // [2*DIM]
                              float* __restrict__ xt,                 // [GM][DIM]
                              float* __restrict__ fo,                 // [GM][DIM]
                              float* __restrict__ ro) {               // [GM][DIM]
  __shared__ __align__(16) unsigned short Als[2][BM * LDST];
  __shared__ __align__(16) unsigned short Bls[2][BN * LDST];

  const int tid  = threadIdx.x;
  const int lane = tid & 31;
  const int wid  = tid >> 5;            // 8 waves
  const int wm   = wid >> 2;            // 0..1 -> 64-row slab
  const int wn   = wid & 3;             // 0..3 -> 32-col slab
  const int mBase = blockIdx.y * BM;
  const int nBase = blockIdx.x * BN;

  v8f acc[4][2];
#pragma unroll
  for (int i = 0; i < 4; ++i)
#pragma unroll
    for (int j = 0; j < 2; ++j)
      acc[i][j] = (v8f)(0.0f);

  // Tile loader: thread owns 16 bf16 of one tile row -> 2x async b128 into LDS
  // (A and B tiles are both 128 rows x 32 k with row length GK in memory)
  const int tr = tid >> 1;              // 0..127 tile row
  const int tk = (tid & 1) * 16;        // 0 or 16
  const unsigned aRowByte = (unsigned)((size_t)(mBase + tr) * GK * 2);  // <= 64MB
  const unsigned bRowByte = (unsigned)((size_t)(nBase + tr) * GK * 2);  // <= 25MB

  const int arow   = lane & 15;
  const int achunk = (lane >> 4) * 8;   // K chunk base per ISA A layout
  const int bcol   = lane & 15;
  const int bchunk = (lane >> 4) * 8;   // K chunk base per ISA B layout (matches A)

  // CDNA5 async DMA global -> LDS; INST_OFFSET applies to both addresses.
  auto issueTile = [&](const unsigned short* src, unsigned rowByte, int ktbase,
                       unsigned short* ldsTile) {
    const unsigned voff = rowByte + (unsigned)((ktbase + tk) * 2);
    const unsigned l0   = (unsigned)(uintptr_t)(ldsTile + tr * LDST + tk);
    asm volatile("global_load_async_to_lds_b128 %0, %1, %2\n\t"
                 "global_load_async_to_lds_b128 %0, %1, %2 offset:16"
                 :: "v"(l0), "v"(voff), "s"(src)
                 : "memory");
  };

  // ---- prologue: tile 0 into buffer 0 ----
  issueTile(A,  aRowByte, 0, &Als[0][0]);
  issueTile(Bt, bRowByte, 0, &Bls[0][0]);
  asm volatile("s_wait_asynccnt 0" ::: "memory");
  __syncthreads();

  for (int kt = 0; kt < KITERS; ++kt) {
    const int  cur  = kt & 1;
    const int  nxt  = cur ^ 1;
    const bool more = (kt + 1 < KITERS);

    if (more) {                          // DMA for next tile overlaps WMMA below
      const int kn = (kt + 1) * BK;
      issueTile(A,  aRowByte, kn, &Als[nxt][0]);
      issueTile(Bt, bRowByte, kn, &Bls[nxt][0]);
    }

    // ---- fragments from current buffer ----
    v16bf afrag[4], bfrag[2];
#pragma unroll
    for (int i = 0; i < 4; ++i) {
      const unsigned short* p = &Als[cur][(wm * 64 + i * 16 + arow) * LDST + achunk];
      v8bf lo = *(const v8bf*)(p);
      v8bf hi = *(const v8bf*)(p + 16);
      afrag[i] = __builtin_shufflevector(lo, hi, 0,1,2,3,4,5,6,7,8,9,10,11,12,13,14,15);
    }
#pragma unroll
    for (int j = 0; j < 2; ++j) {
      const unsigned short* p = &Bls[cur][(wn * 32 + j * 16 + bcol) * LDST + bchunk];
      v8bf lo = *(const v8bf*)(p);
      v8bf hi = *(const v8bf*)(p + 16);
      bfrag[j] = __builtin_shufflevector(lo, hi, 0,1,2,3,4,5,6,7,8,9,10,11,12,13,14,15);
    }

    // ---- 8 WMMAs per wave per K-step ----
#pragma unroll
    for (int i = 0; i < 4; ++i)
#pragma unroll
      for (int j = 0; j < 2; ++j)
        acc[i][j] = __builtin_amdgcn_wmma_f32_16x16x32_bf16(
            false, afrag[i], false, bfrag[j], (short)0, acc[i][j], false, false);

    if (more) asm volatile("s_wait_asynccnt 0" ::: "memory");
    __syncthreads();
  }

  // ---- epilogue: de-interleave N%3, bias+sigmoid for f/r (hoisted per j) --
  const int erow = (lane >> 4) * 8;     // C/D layout: VGPR v -> M = v (+8 for hi lanes)
  const int ecol = lane & 15;
#pragma unroll
  for (int j = 0; j < 2; ++j) {
    const int col  = nBase + wn * 32 + j * 16 + ecol;
    const int role = col % 3;           // 2 divisions per thread total
    const int dd   = col / 3;
    const float bval = (role == 1) ? bias[dd] : ((role == 2) ? bias[DIM + dd] : 0.0f);
    float* outp = (role == 0) ? xt : ((role == 1) ? fo : ro);
    float* base = outp + (size_t)(mBase + wm * 64 + erow) * DIM + dd;
#pragma unroll
    for (int i = 0; i < 4; ++i) {
#pragma unroll
      for (int v = 0; v < 8; ++v) {
        const float val = acc[i][j][v];
        const float sg  = fast_sigmoid(val + bval);
        const float o   = (role == 0) ? val : sg;   // v_cndmask, branch-free
        base[(size_t)(i * 16 + v) * DIM] = o;
      }
    }
  }
}

// ---------------- sequential scan over L, one thread per (b,d) -------------
__global__ void sru_scan(const float* __restrict__ xt,
                         const float* __restrict__ f,
                         const float* __restrict__ r,
                         const float* __restrict__ x,
                         const float* __restrict__ init,
                         float* __restrict__ h,
                         float* __restrict__ c_last) {
  const int bd = blockIdx.x * blockDim.x + threadIdx.x;   // b*DIM + d
  if (bd >= BATCH * DIM) return;
  float c = init[bd];
#pragma unroll 4
  for (int l = 0; l < SEQ; ++l) {
    const size_t idx = (size_t)l * (BATCH * DIM) + bd;
    const float xtv = __builtin_nontemporal_load(xt + idx);  // read-once streams
    const float fv  = __builtin_nontemporal_load(f + idx);
    const float rv  = __builtin_nontemporal_load(r + idx);
    const float xv  = x[idx];
    c = (c - xtv) * fv + xtv;
    const float hv = (fast_tanh(c) - xv) * rv + xv;
    __builtin_nontemporal_store(hv, h + idx);
  }
  c_last[bd] = c;
}

// ---------------------------------------------------------------------------
extern "C" void kernel_launch(void* const* d_in, const int* in_sizes, int n_in,
                              void* d_out, int out_size, void* d_ws, size_t ws_size,
                              hipStream_t stream) {
  const float* x    = (const float*)d_in[0];   // [SEQ][BATCH][DIM]
  const float* W    = (const float*)d_in[1];   // [DIM][3*DIM]
  const float* bias = (const float*)d_in[2];   // [2*DIM]
  const float* init = (const float*)d_in[3];   // [BATCH][DIM]

  float* h_out = (float*)d_out;                          // SEQ*BATCH*DIM
  float* c_out = h_out + (size_t)SEQ * BATCH * DIM;      // BATCH*DIM

  // workspace layout
  char* ws = (char*)d_ws;
  unsigned short* xbf  = (unsigned short*)ws;                         // GM*GK bf16 = 64MB
  unsigned short* wtbf = (unsigned short*)(ws + (size_t)GM * GK * 2); // GN*GK bf16 = 24MB
  float* xt = (float*)(ws + (size_t)GM * GK * 2 + (size_t)GK * GN * 2);
  float* fg = xt + (size_t)GM * DIM;
  float* rg = fg + (size_t)GM * DIM;

  // 1) convert x to bf16; convert+transpose W to Wt[N][K]
  {
    const int n4x = (GM * GK) / 4;
    f32_to_bf16_x4<<<(n4x + 255) / 256, 256, 0, stream>>>(x, xbf, n4x);
    dim3 tg(GN / TT, GK / TT);   // (96, 32)
    w_convert_transpose<<<tg, 256, 0, stream>>>(W, wtbf);
  }

  // 2) WMMA GEMM + fused gates (pure async-DMA data movement)
  {
    dim3 grid(GN / BN, GM / BM);   // (48, 128)
    sru_gemm_bf16<<<grid, 256, 0, stream>>>(xbf, wtbf, bias, xt, fg, rg);
  }

  // 3) scan + h output
  {
    const int n = BATCH * DIM;     // 65536
    sru_scan<<<n / 256, 256, 0, stream>>>(xt, fg, rg, x, init, h_out, c_out);
  }
}
